// WeatherMeshDecoder_5299989643912
// MI455X (gfx1250) — matrix operand order repaired and checked by reading, not compiled, not run on hardware
//
#include <hip/hip_runtime.h>
#include <hip/hip_bf16.h>

// ---------------------------------------------------------------------------
// CDNA5 / gfx1250 WMMA implementation of the WeatherMesh decoder.
// All dense contractions (qkv, proj, split, transposed-conv upsamples) run
// through v_wmma_f32_16x16x32_f16 (f16 A/B, f32 accumulate). Each wave
// computes a 16Mx64N strip (4 accumulators sharing one A fragment); a block
// of 4 waves covers 64Mx64N. The K-loop is software-pipelined with two
// explicit tile buffers so global loads overlap WMMA issue. The 3x5x5
// neighborhood attention (tiny FLOP count) uses VALU online-softmax.
// ---------------------------------------------------------------------------

typedef __attribute__((ext_vector_type(16))) _Float16 v16h;
typedef __attribute__((ext_vector_type(8)))  float    v8f;

union FragAB { v16h v; uint4 q[2]; };

struct Tile {
    FragAB a;
    FragAB b0, b1, b2, b3;
};

__device__ __forceinline__ void load_tile(Tile& t, const _Float16* Ap,
                                          const _Float16* const* Bp, int k0)
{
    t.a.q[0]  = *(const uint4*)(Ap + k0);          // k = k0 + half*16 + [0..7]
    t.a.q[1]  = *(const uint4*)(Ap + k0 + 8);      // k = k0 + half*16 + [8..15]
    t.b0.q[0] = *(const uint4*)(Bp[0] + k0);       // k = k0 + half*16 + [0..7]
    t.b0.q[1] = *(const uint4*)(Bp[0] + k0 + 8);   // k = k0 + half*16 + [8..15]
    t.b1.q[0] = *(const uint4*)(Bp[1] + k0);
    t.b1.q[1] = *(const uint4*)(Bp[1] + k0 + 8);
    t.b2.q[0] = *(const uint4*)(Bp[2] + k0);
    t.b2.q[1] = *(const uint4*)(Bp[2] + k0 + 8);
    t.b3.q[0] = *(const uint4*)(Bp[3] + k0);
    t.b3.q[1] = *(const uint4*)(Bp[3] + k0 + 8);
}

__device__ __forceinline__ void mma_tile(const Tile& t, v8f c[4])
{
    c[0] = __builtin_amdgcn_wmma_f32_16x16x32_f16(
               false, t.a.v, false, t.b0.v, (short)0, c[0], false, false);
    c[1] = __builtin_amdgcn_wmma_f32_16x16x32_f16(
               false, t.a.v, false, t.b1.v, (short)0, c[1], false, false);
    c[2] = __builtin_amdgcn_wmma_f32_16x16x32_f16(
               false, t.a.v, false, t.b2.v, (short)0, c[2], false, false);
    c[3] = __builtin_amdgcn_wmma_f32_16x16x32_f16(
               false, t.a.v, false, t.b3.v, (short)0, c[3], false, false);
}

// ---------------------------------------------------------------------------
// WMMA GEMM:  out[m,n] = sum_k A[m,k] * BT[n,k] + bias[n]
//   A  : [M, K]  f16 row-major   (M % 64 == 0, K % 64 == 0)
//   BT : [Nc, K] f16 row-major   (B transposed; columns of B are rows here)
//   gridDim = (M/64, ceil(Nc/64), numOffsets), blockDim = 128 (4 waves)
//   wave w handles M-tile (blockIdx.x*4 + w), all 4 N sub-tiles.
// Out-of-range N columns load a clamped (valid) B row; their results are
// garbage but never stored -> no divergence in the K loop.
// mode: 0 = linear f16 store out[m*Nc+n]
//       1 = 3D up-conv scatter, f16 token-major   (intermediate layers)
//       2 = 3D up-conv scatter, f32 channel-first (final layer -> d_out)
//       3 = 2D up-conv scatter, f16 token-major
//       4 = 2D up-conv scatter, f32 channel-first
// act:  0 = none, 1 = SiLU
// ---------------------------------------------------------------------------
__global__ __launch_bounds__(128) void gemm64_wmma(
    const _Float16* __restrict__ A,
    const _Float16* __restrict__ BT,
    const float*    __restrict__ bias,
    _Float16*       __restrict__ outH,
    float*          __restrict__ outF,
    int M, int Nc, int K,
    int act, int mode, int Din, int Hin, int Win)
{
    const int lane    = threadIdx.x & 31;   // wave32 lane
    const int wav     = threadIdx.x >> 5;   // 0..3 -> M sub-tile
    const int tm      = blockIdx.x * 4 + wav;
    const int tnBase  = blockIdx.y * 64;
    const int z       = blockIdx.z;         // up-conv offset index
    const int colL    = lane & 15;
    const int halfSel = lane >> 4;

    const _Float16* Bz = BT + (size_t)z * (size_t)Nc * (size_t)K;

    // A fragment: lane holds row (lane&15); k 0..15 / 16..31 per lane>>4
    const int rowA = tm * 16 + colL;
    const _Float16* Ap = A + (size_t)rowA * K + halfSel * 16;

    // B fragments for the 4 N sub-tiles (clamped row for OOB columns)
    const _Float16* Bp[4];
    v8f c[4];
    #pragma unroll
    for (int t = 0; t < 4; ++t) {
        const int cg = tnBase + t * 16 + colL;
        const int cc = (cg < Nc) ? cg : (Nc - 1);
        Bp[t] = Bz + (size_t)cc * K + halfSel * 16;
        const float bv = (bias && cg < Nc) ? bias[cg] : 0.0f;
        c[t] = (v8f){ bv, bv, bv, bv, bv, bv, bv, bv };
    }

    // ---- software-pipelined K loop (K % 64 == 0): two tile buffers ----
    Tile t0, t1;
    load_tile(t0, Ap, Bp, 0);
    int k0 = 0;
    for (; k0 < K - 64; k0 += 64) {
        load_tile(t1, Ap, Bp, k0 + 32);      // overlap with t0 WMMAs
        __builtin_prefetch((const void*)(Ap + k0 + 256), 0, 1);
        mma_tile(t0, c);
        load_tile(t0, Ap, Bp, k0 + 64);      // overlap with t1 WMMAs
        mma_tile(t1, c);
    }
    load_tile(t1, Ap, Bp, k0 + 32);
    mma_tile(t0, c);
    mma_tile(t1, c);

    // ---- epilogue: activation + scatter stores ----
    #pragma unroll
    for (int t = 0; t < 4; ++t) {
        const int colg = tnBase + t * 16 + colL;
        if (colg >= Nc) continue;
        #pragma unroll
        for (int r = 0; r < 8; ++r) {
            float v = c[t][r];
            if (act == 1) v = v / (1.0f + __expf(-v));   // SiLU
            const int rowT = tm * 16 + r + 8 * halfSel;  // C/D: vgpr r -> row r / r+8

            if (mode == 0) {
                outH[(size_t)rowT * Nc + colg] = (_Float16)v;
            } else if (mode == 1 || mode == 2) {
                const int oi = (z >> 2) & 1, oj = (z >> 1) & 1, ok = z & 1;
                const int n  = rowT;
                const int dd = n / (Hin * Win);
                const int rm = n - dd * Hin * Win;
                const int hh = rm / Win;
                const int ww = rm - hh * Win;
                const long np = ((long)(2 * dd + oi) * (2 * Hin) + (2 * hh + oj))
                                    * (2 * Win) + (2 * ww + ok);
                if (mode == 1) outH[(size_t)np * Nc + colg] = (_Float16)v;
                else           outF[(size_t)colg * (8LL * Din * Hin * Win) + np] = v;
            } else {
                const int oi = (z >> 1) & 1, oj = z & 1;
                const int n  = rowT;
                const int hh = n / Win;
                const int ww = n - hh * Win;
                const long np = (long)(2 * hh + oi) * (2 * Win) + (2 * ww + oj);
                if (mode == 3) outH[(size_t)np * Nc + colg] = (_Float16)v;
                else           outF[(size_t)colg * (4LL * Hin * Win) + np] = v;
            }
        }
    }
}

// ---------------------------------------------------------------------------
// 3x5x5 neighborhood attention, one thread per (token, head), online softmax.
// qkv: [Ntok, 768] f16 (q|k|v), out: [Ntok, 256] f16
// ---------------------------------------------------------------------------
__global__ __launch_bounds__(256) void na3d_attn(
    const _Float16* __restrict__ qkv, _Float16* __restrict__ outp,
    int D, int H, int W)
{
    const int gid = blockIdx.x * blockDim.x + threadIdx.x;
    const int total = D * H * W * 8;
    if (gid >= total) return;
    const int head = gid & 7;
    const int tok  = gid >> 3;
    const int w = tok % W;
    const int h = (tok / W) % H;
    const int d = tok / (W * H);

    const _Float16* qp = qkv + (size_t)tok * 768 + head * 32;
    float q[32];
    #pragma unroll
    for (int i = 0; i < 32; ++i) q[i] = (float)qp[i];

    float m = -1e30f, s = 0.0f, acc[32];
    #pragma unroll
    for (int i = 0; i < 32; ++i) acc[i] = 0.0f;

    const int ds  = min(max(d - 1, 0), D - 3);
    const int hs  = min(max(h - 2, 0), H - 5);
    const int ws0 = min(max(w - 2, 0), W - 5);

    for (int cd = ds; cd < ds + 3; ++cd)
    for (int ch = hs; ch < hs + 5; ++ch)
    for (int cw = ws0; cw < ws0 + 5; ++cw) {
        const int t = (cd * H + ch) * W + cw;
        const _Float16* kp = qkv + (size_t)t * 768 + 256 + head * 32;
        float sc = 0.0f;
        #pragma unroll
        for (int i = 0; i < 32; ++i) sc += q[i] * (float)kp[i];
        sc *= 0.17677669529663687f;            // 1/sqrt(32)
        const float nm  = fmaxf(m, sc);
        const float f   = __expf(m - nm);
        const float wgt = __expf(sc - nm);
        s = s * f + wgt;
        const _Float16* vp = kp + 256;
        #pragma unroll
        for (int i = 0; i < 32; ++i) acc[i] = acc[i] * f + wgt * (float)vp[i];
        m = nm;
    }

    const float inv = 1.0f / s;
    _Float16* op = outp + (size_t)tok * 256 + head * 32;
    #pragma unroll
    for (int i = 0; i < 32; ++i) op[i] = (_Float16)(acc[i] * inv);
}

// ---------------------------------------------------------------------------
// Operand preparation kernels (f32 -> f16, with layout changes for B^T)
// ---------------------------------------------------------------------------
__global__ void cvt_f16(const float* __restrict__ src, _Float16* __restrict__ dst, long n)
{
    const long i = (long)blockIdx.x * blockDim.x + threadIdx.x;
    if (i < n) dst[i] = (_Float16)src[i];
}

// W [K, N] f32 -> WT [N, K] f16
__global__ void transpose_cvt(const float* __restrict__ W, _Float16* __restrict__ WT,
                              int K, int N)
{
    const long i = (long)blockIdx.x * blockDim.x + threadIdx.x;
    if (i >= (long)K * N) return;
    const int n = (int)(i / K);
    const int k = (int)(i % K);
    WT[i] = (_Float16)W[(size_t)k * N + n];
}

// ConvTranspose weight (C, O, off...) f32 -> (noff, O, C) f16
__global__ void convw_cvt(const float* __restrict__ W, _Float16* __restrict__ WT,
                          int C, int O, int noff)
{
    const long total = (long)noff * O * C;
    const long i = (long)blockIdx.x * blockDim.x + threadIdx.x;
    if (i >= total) return;
    const int off = (int)(i / ((long)O * C));
    const int rem = (int)(i % ((long)O * C));
    const int o   = rem / C;
    const int cc  = rem - o * C;
    WT[i] = (_Float16)W[((size_t)cc * O + o) * noff + off];
}

// ---------------------------------------------------------------------------
// Host orchestration
// ---------------------------------------------------------------------------
extern "C" void kernel_launch(void* const* d_in, const int* in_sizes, int n_in,
                              void* d_out, int out_size, void* d_ws, size_t ws_size,
                              hipStream_t stream)
{
    (void)in_sizes; (void)n_in; (void)out_size; (void)ws_size;

    const float* latent  = (const float*)d_in[0];   // (1,5,24,48,256)
    const float* qkv_w   = (const float*)d_in[1];   // (3,256,768)
    const float* qkv_b   = (const float*)d_in[2];   // (3,768)
    const float* proj_w  = (const float*)d_in[3];   // (3,256,256)
    const float* proj_b  = (const float*)d_in[4];   // (3,256)
    const float* split_w = (const float*)d_in[5];   // (512,256)  == B^T already
    const float* split_b = (const float*)d_in[6];   // (512,)
    const float* pw0 = (const float*)d_in[7];  const float* pb0 = (const float*)d_in[8];
    const float* pw1 = (const float*)d_in[9];  const float* pb1 = (const float*)d_in[10];
    const float* pw2 = (const float*)d_in[11]; const float* pb2 = (const float*)d_in[12];
    const float* sw0 = (const float*)d_in[13]; const float* sb0 = (const float*)d_in[14];
    const float* sw1 = (const float*)d_in[15]; const float* sb1 = (const float*)d_in[16];
    const float* sw2 = (const float*)d_in[17]; const float* sb2 = (const float*)d_in[18];

    float* outF = (float*)d_out;   // surf (8*96*192) then press (5*32*192*384)
    const long SURF_ELEMS = 8L * 96 * 192;          // 147456

    // ---- workspace bump allocator (256B aligned) ----
    char* wsb = (char*)d_ws;
    size_t cur = 0;
    auto alloc = [&](size_t bytes) -> void* {
        size_t p = (cur + 255) & ~(size_t)255;
        cur = p + bytes;
        return (void*)(wsb + p);
    };

    const int NTOK = 5 * 24 * 48;                   // 5760 tokens

    _Float16* h_x    = (_Float16*)alloc((size_t)NTOK * 256 * 2);
    _Float16* h_qkv  = (_Float16*)alloc((size_t)NTOK * 768 * 2);
    _Float16* h_attn = (_Float16*)alloc((size_t)NTOK * 256 * 2);
    _Float16* h_feat = (_Float16*)alloc((size_t)NTOK * 512 * 2);

    _Float16* wt_qkv  = (_Float16*)alloc((size_t)3 * 768 * 256 * 2);
    _Float16* wt_proj = (_Float16*)alloc((size_t)3 * 256 * 256 * 2);
    _Float16* h_splw  = (_Float16*)alloc((size_t)512 * 256 * 2);
    _Float16* wt_p0   = (_Float16*)alloc((size_t)8 * 256 * 512 * 2);
    _Float16* wt_p1   = (_Float16*)alloc((size_t)8 * 128 * 256 * 2);
    _Float16* wt_p2   = (_Float16*)alloc((size_t)8 * 5 * 128 * 2);
    _Float16* wt_s0   = (_Float16*)alloc((size_t)4 * 256 * 512 * 2);
    _Float16* wt_s1   = (_Float16*)alloc((size_t)4 * 128 * 256 * 2);
    _Float16* wt_s2   = (_Float16*)alloc((size_t)4 * 8 * 128 * 2);

    _Float16* p0out = (_Float16*)alloc((size_t)36864 * 256 * 2);
    _Float16* p1out = (_Float16*)alloc((size_t)294912 * 128 * 2);
    _Float16* s0out = (_Float16*)alloc((size_t)4608 * 256 * 2);
    _Float16* s1out = (_Float16*)alloc((size_t)18432 * 128 * 2);

    auto blocksFor = [](long n) { return (unsigned)((n + 255) / 256); };
    auto ceil64    = [](int n)  { return (unsigned)((n + 63) / 64); };

    // ---- operand prep ----
    cvt_f16<<<blocksFor((long)NTOK * 256), 256, 0, stream>>>(latent, h_x, (long)NTOK * 256);
    cvt_f16<<<blocksFor(512L * 256), 256, 0, stream>>>(split_w, h_splw, 512L * 256);
    for (int l = 0; l < 3; ++l) {
        transpose_cvt<<<blocksFor(768L * 256), 256, 0, stream>>>(
            qkv_w + (size_t)l * 256 * 768, wt_qkv + (size_t)l * 768 * 256, 256, 768);
        transpose_cvt<<<blocksFor(256L * 256), 256, 0, stream>>>(
            proj_w + (size_t)l * 256 * 256, wt_proj + (size_t)l * 256 * 256, 256, 256);
    }
    convw_cvt<<<blocksFor(8L * 256 * 512), 256, 0, stream>>>(pw0, wt_p0, 512, 256, 8);
    convw_cvt<<<blocksFor(8L * 128 * 256), 256, 0, stream>>>(pw1, wt_p1, 256, 128, 8);
    convw_cvt<<<blocksFor(8L * 5 * 128),   256, 0, stream>>>(pw2, wt_p2, 128, 5, 8);
    convw_cvt<<<blocksFor(4L * 256 * 512), 256, 0, stream>>>(sw0, wt_s0, 512, 256, 4);
    convw_cvt<<<blocksFor(4L * 128 * 256), 256, 0, stream>>>(sw1, wt_s1, 256, 128, 4);
    convw_cvt<<<blocksFor(4L * 8 * 128),   256, 0, stream>>>(sw2, wt_s2, 128, 8, 4);

    const dim3 blk(128);   // 4 waves

    // ---- 3x neighborhood-attention blocks ----
    for (int l = 0; l < 3; ++l) {
        // qkv = x @ Wqkv + b   [5760,256] x [256,768]
        gemm64_wmma<<<dim3(NTOK / 64, ceil64(768), 1), blk, 0, stream>>>(
            h_x, wt_qkv + (size_t)l * 768 * 256, qkv_b + (size_t)l * 768,
            h_qkv, nullptr, NTOK, 768, 256, 0, 0, 0, 0, 0);
        // attention
        na3d_attn<<<blocksFor((long)NTOK * 8), 256, 0, stream>>>(h_qkv, h_attn, 5, 24, 48);
        // x = attn @ Wproj + b  [5760,256] x [256,256]
        gemm64_wmma<<<dim3(NTOK / 64, ceil64(256), 1), blk, 0, stream>>>(
            h_attn, wt_proj + (size_t)l * 256 * 256, proj_b + (size_t)l * 256,
            h_x, nullptr, NTOK, 256, 256, 0, 0, 0, 0, 0);
    }

    // ---- split head: feats[n,o] = x[n,:] . split_w[o,:] + b ----
    gemm64_wmma<<<dim3(NTOK / 64, ceil64(512), 1), blk, 0, stream>>>(
        h_x, h_splw, split_b, h_feat, nullptr, NTOK, 512, 256, 0, 0, 0, 0, 0);

    // Token layout is (d,h,w)-major: press = rows [0,4608), surf = rows [4608,5760)
    _Float16* pressA = h_feat;
    _Float16* surfA  = h_feat + (size_t)4608 * 512;

    // ---- pressure path: three 3D stride-2 transposed convs ----
    gemm64_wmma<<<dim3(4608 / 64, ceil64(256), 8), blk, 0, stream>>>(
        pressA, wt_p0, pb0, p0out, nullptr, 4608, 256, 512, 1, 1, 4, 24, 48);
    gemm64_wmma<<<dim3(36864 / 64, ceil64(128), 8), blk, 0, stream>>>(
        p0out, wt_p1, pb1, p1out, nullptr, 36864, 128, 256, 1, 1, 8, 48, 96);
    gemm64_wmma<<<dim3(294912 / 64, 1, 8), blk, 0, stream>>>(
        p1out, wt_p2, pb2, nullptr, outF + SURF_ELEMS, 294912, 5, 128, 0, 2, 16, 96, 192);

    // ---- surface path: three 2D stride-2 transposed convs ----
    gemm64_wmma<<<dim3(1152 / 64, ceil64(256), 4), blk, 0, stream>>>(
        surfA, wt_s0, sb0, s0out, nullptr, 1152, 256, 512, 1, 3, 0, 24, 48);
    gemm64_wmma<<<dim3(4608 / 64, ceil64(128), 4), blk, 0, stream>>>(
        s0out, wt_s1, sb1, s1out, nullptr, 4608, 128, 256, 1, 3, 0, 48, 96);
    gemm64_wmma<<<dim3(18432 / 64, 1, 4), blk, 0, stream>>>(
        s1out, wt_s2, sb2, nullptr, outF, 18432, 8, 128, 0, 4, 0, 96, 192);
}
